// GraphSage_3246995276246
// MI455X (gfx1250) — matrix operand order repaired and checked by reading, not compiled, not run on hardware
//
#include <hip/hip_runtime.h>

typedef __attribute__((ext_vector_type(16))) _Float16 v16h;
typedef __attribute__((ext_vector_type(8)))  _Float16 v8h;
typedef __attribute__((ext_vector_type(4)))  _Float16 v4h;
typedef __attribute__((ext_vector_type(8)))  float    v8f;
typedef __attribute__((ext_vector_type(4)))  float    v4f;

namespace {
constexpr int kF       = 128;   // raw feature dim (layer-1 input)
constexpr int kD       = 128;   // embedding dim
constexpr int kK       = 256;   // concat dim = 2*128 (GEMM K)
constexpr int kS       = 10;    // neighbor fanout
constexpr int kRows    = 128;   // rows per workgroup
constexpr int kThreads = 256;   // 8 waves (wave32)
constexpr int kN1      = 81920;
constexpr int kB       = 8192;
// Row stride in halfs for LDS tiles: 264 halfs = 132 dwords; 132 % 64 == 4, so
// 16 consecutive rows map to disjoint 4-dword bank groups (conflict-free b128).
constexpr int kStride  = 264;
constexpr size_t kSmemBytes = (size_t)(2 * kRows) * kStride * sizeof(_Float16); // 132 KB
}

union AFrag { v16h v; v8h h[2]; };

// Cooperative load of W [128 x 256] fp32 -> f16 LDS tile with padded rows.
__device__ __forceinline__ void load_weights_f16(const float* __restrict__ W,
                                                 _Float16* __restrict__ Wlds,
                                                 int tid)
{
#pragma unroll
    for (int i = 0; i < 32; ++i) {                // 8192 float4 / 256 threads
        const int idx4 = i * kThreads + tid;
        const int row  = idx4 >> 6;               // flat/256
        const int col  = (idx4 & 63) << 2;
        v4f w = *(const v4f*)(W + idx4 * 4);
        *(v4h*)(Wlds + row * kStride + col) = __builtin_convertvector(w, v4h);
    }
}

// ---------------------------------------------------------------------------
// Layer 1: gather fp32 raw features, mean-aggregate, f16 WMMA GEMM, relu,
// write h1 as f16 into workspace.
// ---------------------------------------------------------------------------
__global__ __launch_bounds__(kThreads)
void sage_layer1_kernel(const float* __restrict__ raw,
                        const float* __restrict__ W1,
                        const int*   __restrict__ selfIdx,
                        const int*   __restrict__ neighIdx,
                        _Float16*    __restrict__ h1)
{
    extern __shared__ _Float16 smem[];
    _Float16* comb = smem;                        // [128][264] (256 used)
    _Float16* Wlds = smem + kRows * kStride;      // [128][264] (256 used)
    const int tid = threadIdx.x;
    const int wg  = blockIdx.x;

    load_weights_f16(W1, Wlds, tid);

    // ---- Phase 1: gather self + mean(neighbors) -> f16 LDS tile ----
    {
        const int r    = tid >> 1;                // row within WG
        const int fb   = (tid & 1) * (kF / 2);    // 64-feature half
        const int grow = wg * kRows + r;
        const int self = selfIdx[grow];
        int nb[kS];
#pragma unroll
        for (int j = 0; j < kS; ++j) nb[j] = neighIdx[grow * kS + j];
        const float* srow = raw + (long)self * kF + fb;
        _Float16* crow = comb + r * kStride;
#pragma unroll 4
        for (int i = 0; i < 16; ++i) {            // 16 chunks x 4 floats = 64
            v4f sv = *(const v4f*)(srow + i * 4);
            v4f av = {0.f, 0.f, 0.f, 0.f};
#pragma unroll
            for (int j = 0; j < kS; ++j)
                av += *(const v4f*)(raw + (long)nb[j] * kF + fb + i * 4);
            av *= 0.1f;                           // mean over 10 samples
            *(v4h*)(crow + fb + i * 4)      = __builtin_convertvector(sv, v4h);
            *(v4h*)(crow + kF + fb + i * 4) = __builtin_convertvector(av, v4h);
        }
    }
    __syncthreads();

    // ---- Phase 2: per-wave 16-row x 128-col WMMA GEMM ----
    const int lane    = tid & 31;
    const int wave    = tid >> 5;
    const int rbase   = wave * 16;
    const int laneRow = lane & 15;
    const int hiOff   = (lane >> 4) * 8;          // A K-offset / C-D row offset
    const int hiOffB  = (lane >> 4) * 16;         // unused; B uses hiOff

    AFrag a[8];
    const _Float16* arow = comb + (rbase + laneRow) * kStride;
#pragma unroll
    for (int k = 0; k < 8; ++k) {                 // K = 256 in 8 chunks of 32
        const int kb = k * 32 + hiOff;
        a[k].h[0] = *(const v8h*)(arow + kb);         // K 0-7 / 8-15
        a[k].h[1] = *(const v8h*)(arow + kb + 16);    // K 16-23 / 24-31
    }

    _Float16* orow = comb + rbase * kStride;      // wave-private 16x128 staging
#pragma unroll
    for (int nt = 0; nt < 8; ++nt) {
        const _Float16* wr = Wlds + (nt * 16 + laneRow) * kStride;
        v8f acc = {0.f, 0.f, 0.f, 0.f, 0.f, 0.f, 0.f, 0.f};
#pragma unroll
        for (int k = 0; k < 8; ++k) {
            const int kb = k * 32 + hiOff;        // K 0-7/8-15 and 16-23/24-31, same as A
            AFrag b;
            b.h[0] = *(const v8h*)(wr + kb);
            b.h[1] = *(const v8h*)(wr + kb + 16);
            acc = __builtin_amdgcn_wmma_f32_16x16x32_f16(
                false, a[k].v, false, b.v, (short)0, acc, false, false);
        }
#pragma unroll
        for (int j = 0; j < 8; ++j) {             // relu + f16, scatter to LDS
            float v = acc[j];
            v = v > 0.f ? v : 0.f;
            orow[(j + hiOff) * kD + nt * 16 + laneRow] = (_Float16)v;
        }
    }

    // ---- coalesced writeback of h1 tile (16 x 128 f16) ----
    _Float16* gout = h1 + (long)(wg * kRows + rbase) * kD;
#pragma unroll
    for (int r = 0; r < 16; ++r)
        *(v4h*)(gout + r * kD + lane * 4) =
            *(const v4h*)(orow + r * kD + lane * 4);
}

// ---------------------------------------------------------------------------
// Layer 2: gather f16 h1 rows, mean-aggregate in fp32, f16 WMMA GEMM, relu,
// write fp32 output.
// ---------------------------------------------------------------------------
__global__ __launch_bounds__(kThreads)
void sage_layer2_kernel(const _Float16* __restrict__ h1,
                        const float*    __restrict__ W2,
                        const int*      __restrict__ selfIdx,
                        const int*      __restrict__ neighIdx,
                        float*          __restrict__ out)
{
    extern __shared__ _Float16 smem[];
    _Float16* comb = smem;
    _Float16* Wlds = smem + kRows * kStride;
    const int tid = threadIdx.x;
    const int wg  = blockIdx.x;

    load_weights_f16(W2, Wlds, tid);

    // ---- Phase 1: gather from h1 (f16), mean in fp32 ----
    {
        const int r    = tid >> 1;
        const int fb   = (tid & 1) * (kD / 2);
        const int grow = wg * kRows + r;
        const int self = selfIdx[grow];
        int nb[kS];
#pragma unroll
        for (int j = 0; j < kS; ++j) nb[j] = neighIdx[grow * kS + j];
        _Float16* crow = comb + r * kStride;
        const _Float16* srow = h1 + (long)self * kD + fb;
#pragma unroll
        for (int i = 0; i < 8; ++i)               // self copy: 64 halfs
            *(v8h*)(crow + fb + i * 8) = *(const v8h*)(srow + i * 8);
#pragma unroll 4
        for (int i = 0; i < 16; ++i) {
            v4f av = {0.f, 0.f, 0.f, 0.f};
#pragma unroll
            for (int j = 0; j < kS; ++j) {
                v4h t = *(const v4h*)(h1 + (long)nb[j] * kD + fb + i * 4);
                av += __builtin_convertvector(t, v4f);
            }
            av *= 0.1f;
            *(v4h*)(crow + kD + fb + i * 4) = __builtin_convertvector(av, v4h);
        }
    }
    __syncthreads();

    // ---- Phase 2: WMMA ----
    const int lane    = tid & 31;
    const int wave    = tid >> 5;
    const int rbase   = wave * 16;
    const int laneRow = lane & 15;
    const int hiOff   = (lane >> 4) * 8;
    const int hiOffB  = (lane >> 4) * 16;

    AFrag a[8];
    const _Float16* arow = comb + (rbase + laneRow) * kStride;
#pragma unroll
    for (int k = 0; k < 8; ++k) {
        const int kb = k * 32 + hiOff;
        a[k].h[0] = *(const v8h*)(arow + kb);
        a[k].h[1] = *(const v8h*)(arow + kb + 16);
    }

    // wave-private region: 16*264 halfs >= 16x128 floats of staging
    float* orow = (float*)(comb + rbase * kStride);
#pragma unroll
    for (int nt = 0; nt < 8; ++nt) {
        const _Float16* wr = Wlds + (nt * 16 + laneRow) * kStride;
        v8f acc = {0.f, 0.f, 0.f, 0.f, 0.f, 0.f, 0.f, 0.f};
#pragma unroll
        for (int k = 0; k < 8; ++k) {
            const int kb = k * 32 + hiOff;
            AFrag b;
            b.h[0] = *(const v8h*)(wr + kb);
            b.h[1] = *(const v8h*)(wr + kb + 16);
            acc = __builtin_amdgcn_wmma_f32_16x16x32_f16(
                false, a[k].v, false, b.v, (short)0, acc, false, false);
        }
#pragma unroll
        for (int j = 0; j < 8; ++j) {
            float v = acc[j];
            v = v > 0.f ? v : 0.f;
            orow[(j + hiOff) * kD + nt * 16 + laneRow] = v;
        }
    }

    // ---- coalesced fp32 writeback (16 x 128 f32) ----
    float* gout = out + (long)(wg * kRows + rbase) * kD;
#pragma unroll
    for (int r = 0; r < 16; ++r)
        *(v4f*)(gout + r * kD + lane * 4) =
            *(const v4f*)(orow + r * kD + lane * 4);
}

// ---------------------------------------------------------------------------
extern "C" void kernel_launch(void* const* d_in, const int* in_sizes, int n_in,
                              void* d_out, int out_size, void* d_ws, size_t ws_size,
                              hipStream_t stream) {
    const float* raw       = (const float*)d_in[0];
    const float* W1        = (const float*)d_in[1];
    const float* W2        = (const float*)d_in[2];
    const int*   selfIdx1  = (const int*)d_in[3];
    const int*   neighIdx1 = (const int*)d_in[4];
    const int*   selfIdx2  = (const int*)d_in[5];
    const int*   neighIdx2 = (const int*)d_in[6];
    float*       out       = (float*)d_out;
    _Float16*    h1        = (_Float16*)d_ws;     // 81920*128 f16 = 20 MB

    sage_layer1_kernel<<<kN1 / kRows, kThreads, kSmemBytes, stream>>>(
        raw, W1, selfIdx1, neighIdx1, h1);
    sage_layer2_kernel<<<kB / kRows, kThreads, kSmemBytes, stream>>>(
        h1, W2, selfIdx2, neighIdx2, out);
}
